// GIN_32976758898936
// MI455X (gfx1250) — compile-verified, matrix operand order checked by reading, not run on hardware
//
#include <hip/hip_runtime.h>
#include <hip/hip_bf16.h>

// ---------------- problem constants (from reference) ----------------
#define NN 50000        // nodes
#define NE 800000       // edges
#define SS 2            // samples
#define FIN 96
#define FNOISE 32
#define KD 128          // F_IN + F_NOISE == D0 == D1
#define MM (SS * NN)    // 100000 rows, divisible by 16 (6250 strips)

typedef _Float16 half16 __attribute__((ext_vector_type(16)));
typedef float    float8 __attribute__((ext_vector_type(8)));

// ---------------- concat(X, epsilon) -> x0 fp32 [S,N,128] ----------------
__global__ __launch_bounds__(256) void gin_concat(const float* __restrict__ X,
                                                  const float* __restrict__ Eps,
                                                  float* __restrict__ x0) {
    long i = (long)blockIdx.x * 256 + threadIdx.x;   // over S*N*128
    if (i >= (long)MM * KD) return;
    int  f  = (int)(i & (KD - 1));
    long sn = i >> 7;
    x0[i] = (f < FIN) ? X[sn * FIN + f] : Eps[sn * FNOISE + (f - FIN)];
}

// ---------------- edge scatter: agg[dst] += x[src] (L2-resident atomics) ----------------
__global__ __launch_bounds__(256) void gin_scatter(const int* __restrict__ src,
                                                   const int* __restrict__ dst,
                                                   const float* __restrict__ x,
                                                   float* __restrict__ agg) {
    long t = (long)blockIdx.x * 256 + threadIdx.x;   // E * 32 lanes
    if (t >= (long)NE * 32) return;
    int e  = (int)(t >> 5);
    int f4 = (int)(t & 31) << 2;                      // 4 floats per lane
    int u = src[e], v = dst[e];
#pragma unroll
    for (int s = 0; s < SS; ++s) {
        const float4 xv = *(const float4*)&x[((size_t)s * NN + u) * KD + f4];
        float* ap = &agg[((size_t)s * NN + v) * KD + f4];
        atomicAdd(ap + 0, xv.x);
        atomicAdd(ap + 1, xv.y);
        atomicAdd(ap + 2, xv.z);
        atomicAdd(ap + 3, xv.w);
    }
}

// ---------------- a16 = f16((1+eps)*x + agg) ----------------
__global__ __launch_bounds__(256) void gin_prep(const float* __restrict__ x,
                                                const float* __restrict__ agg,
                                                const float* __restrict__ epsp,
                                                _Float16* __restrict__ a16) {
    long i = (long)blockIdx.x * 256 + threadIdx.x;   // over M*128/4
    if (i >= (long)MM * KD / 4) return;
    const float g = 1.0f + epsp[0];
    float4 xv = ((const float4*)x)[i];
    float4 av = ((const float4*)agg)[i];
    _Float16 h4[4];
    h4[0] = (_Float16)(g * xv.x + av.x);
    h4[1] = (_Float16)(g * xv.y + av.y);
    h4[2] = (_Float16)(g * xv.z + av.z);
    h4[3] = (_Float16)(g * xv.w + av.w);
    *(uint2*)&a16[4 * i] = *(uint2*)h4;              // 8B store
}

// ---------------- pack W fp32[128,128] row-major -> WMMA 16-bit B-operand layout ----------------
// Layout per ISA 7.12.2 (16-bit operand, 16x32 per k-step):
//   lane L: col n = nt*16 + (L&15), h = L>>4
//   element e (0..15): k_local = e + 8h + (e>=8 ? 8 : 0), k = 32*kb + k_local
// Packed so each lane reads 16 contiguous halfs (32B) per k-step.
__global__ __launch_bounds__(256) void gin_packw(const float* __restrict__ Wf,
                                                 _Float16* __restrict__ Wp) {
    int t = blockIdx.x * 256 + threadIdx.x;          // 8 nt * 4 kb * 32 lanes * 16 e = 16384
    if (t >= 8 * 4 * 32 * 16) return;
    int e  = t & 15;
    int L  = (t >> 4) & 31;
    int kb = (t >> 9) & 3;
    int nt = t >> 11;
    int h  = L >> 4;
    int kl = e + 8 * h + (e >= 8 ? 8 : 0);
    int k  = 32 * kb + kl;
    int n  = nt * 16 + (L & 15);
    Wp[t] = (_Float16)Wf[k * KD + n];
}

// ---------------- GEMM: Out = relu(A @ W + b), A f16 [M,128], W packed, Out f32 [M,128] ----------------
// One block per 16-row strip; 8 waves, wave w owns column tile nt=w; K=128 -> 4x wmma_f32_16x16x32_f16.
__global__ __launch_bounds__(256) void gin_gemm_wmma(const _Float16* __restrict__ A,
                                                     const _Float16* __restrict__ Wp,
                                                     const float* __restrict__ bias,
                                                     float* __restrict__ Out) {
    __shared__ _Float16 As[16 * KD];                 // 4 KB strip
    const int row0 = blockIdx.x * 16;
    const int tid  = threadIdx.x;

    // cooperative 4KB strip load: 256 threads x 16B
    ((uint4*)As)[tid] = ((const uint4*)(A + (size_t)row0 * KD))[tid];
    __syncthreads();

    const int lane = tid & 31;
    const int nt   = tid >> 5;                        // 0..7 column tile
    const int m    = lane & 15;                       // A row / C column within tile
    const int h    = lane >> 4;                       // half-wave select

    float8 c = {};
#pragma unroll
    for (int kb = 0; kb < 4; ++kb) {
        // A operand, per ISA 16-bit A 16x32 layout: e<8 -> k = e+8h ; e>=8 -> k = e+8+8h
        const _Float16* arow = &As[m * KD + kb * 32 + 8 * h];
        half16 a, b;
#pragma unroll
        for (int e = 0; e < 8; ++e)  a[e] = arow[e];
#pragma unroll
        for (int e = 8; e < 16; ++e) a[e] = arow[e + 8];
        // B operand: 32B contiguous per lane from the packed buffer
        b = *(const half16*)(Wp + (((size_t)nt * 4 + kb) * 32 + lane) * 16);
        c = __builtin_amdgcn_wmma_f32_16x16x32_f16(false, a, false, b, (short)0, c,
                                                   false, false);
    }

    // C/D layout: lane -> column (lane&15), VGPR j -> row j + 8*(lane>=16)
    const float bn = bias[nt * 16 + m];
#pragma unroll
    for (int j = 0; j < 8; ++j) {
        int r = row0 + j + 8 * h;
        float v = c[j] + bn;
        Out[(size_t)r * KD + nt * 16 + m] = v > 0.0f ? v : 0.0f;
    }
}

// ---------------- host orchestration ----------------
extern "C" void kernel_launch(void* const* d_in, const int* in_sizes, int n_in,
                              void* d_out, int out_size, void* d_ws, size_t ws_size,
                              hipStream_t stream) {
    const int*   src  = (const int*)d_in[0];
    const int*   dst  = (const int*)d_in[1];
    const float* X    = (const float*)d_in[2];
    const float* Eps  = (const float*)d_in[3];
    const float* W1   = (const float*)d_in[4];
    const float* b1   = (const float*)d_in[5];
    const float* W2   = (const float*)d_in[6];
    const float* b2   = (const float*)d_in[7];
    const float* eps0 = (const float*)d_in[8];
    const float* eps1 = (const float*)d_in[9];
    float* out = (float*)d_out;

    // workspace carve-up (needs ~128.1 MB; all offsets 16B-aligned)
    char* ws = (char*)d_ws;
    float*    agg = (float*)(ws);                          // 51,200,000 B
    float*    x0  = (float*)(ws + 51200000);               // 51,200,000 B
    _Float16* a16 = (_Float16*)(ws + 102400000);           // 25,600,000 B
    _Float16* wp1 = (_Float16*)(ws + 128000000);           //     32,768 B
    _Float16* wp2 = (_Float16*)(ws + 128032768);           //     32,768 B
    (void)in_sizes; (void)n_in; (void)out_size; (void)ws_size;

    const size_t aggBytes = (size_t)MM * KD * sizeof(float);

    // layer-invariant prep
    gin_concat<<<(MM * KD + 255) / 256, 256, 0, stream>>>(X, Eps, x0);
    gin_packw<<<16384 / 256, 256, 0, stream>>>(W1, wp1);
    gin_packw<<<16384 / 256, 256, 0, stream>>>(W2, wp2);

    const int scatterBlocks = (int)(((long)NE * 32 + 255) / 256);
    const int prepBlocks    = (int)(((long)MM * KD / 4 + 255) / 256);
    const int gemmBlocks    = MM / 16;

    // ---- layer 1: x0 -> out ----
    hipMemsetAsync(agg, 0, aggBytes, stream);
    gin_scatter<<<scatterBlocks, 256, 0, stream>>>(src, dst, x0, agg);
    gin_prep<<<prepBlocks, 256, 0, stream>>>(x0, agg, eps0, a16);
    gin_gemm_wmma<<<gemmBlocks, 256, 0, stream>>>(a16, wp1, b1, out);

    // ---- layer 2: out -> out ----
    hipMemsetAsync(agg, 0, aggBytes, stream);
    gin_scatter<<<scatterBlocks, 256, 0, stream>>>(src, dst, out, agg);
    gin_prep<<<prepBlocks, 256, 0, stream>>>(out, agg, eps1, a16);
    gin_gemm_wmma<<<gemmBlocks, 256, 0, stream>>>(a16, wp2, b2, out);
    // final relu(relu(x)) == relu(x): already applied in gemm epilogue
}
